// network_481036337395
// MI455X (gfx1250) — matrix operand order repaired and checked by reading, not compiled, not run on hardware
//
#include <hip/hip_runtime.h>
#include <hip/hip_bf16.h>

// ---------------------------------------------------------------------------
// CDNA5 (gfx1250) implementation of the GAT/GNN forward pass.
// Dense GEMMs: v_wmma_f32_16x16x32_f16, wave32, each wave owns a 32x16 output
// block (two 16x16 WMMA tiles sharing one B fragment); A fragments loaded as
// float4 (global_load_b128). Edge softmax / segment ops use global f32 atomics
// + order-preserving uint encoding for float atomic-max. BN column stats use
// LDS atomics for block-local reduction.
// ---------------------------------------------------------------------------

typedef __attribute__((ext_vector_type(16))) _Float16 v16h;
typedef __attribute__((ext_vector_type(8)))  float    v8f;
typedef __attribute__((ext_vector_type(4)))  float    v4f;

#define NRn 200000
#define NDn 100000
#define NSn 100000
#define ERn 800000
#define EIn 800000
#define EIDn 400000
#define EDn 400000
#define ESn 400000
#define HID 64
#define Bn 1024
#define NLn 4

// ---------------- generic fill ----------------
__global__ void fill_u32(unsigned* __restrict__ p, unsigned v, long n) {
  long i = (long)blockIdx.x * blockDim.x + threadIdx.x;
  if (i < n) p[i] = v;
}

// ---------------- WMMA GEMM: Y[N x Co] = act(X[N x Ci] @ W[Ci x Co] + bias) ----
// Each wave computes a 32(M) x 16(N) block = two 16x16 WMMA tiles sharing one
// B fragment. rowidx != null => gather rows of X through rowidx. accumulate =>
// Y += result (split-K over concatenated inputs). act: 0 none, 1 lrelu(0.01),
// 2 relu. Requires Ci % 32 == 0 and Co % 16 == 0 (true for all call sites).
__global__ void gemm_wmma(const float* __restrict__ X, const int* __restrict__ rowidx,
                          const float* __restrict__ W, const float* __restrict__ bias,
                          float* __restrict__ Y, int N, int Ci, int Co,
                          int accumulate, int act) {
  int wave   = blockIdx.x * (blockDim.x >> 5) + (threadIdx.x >> 5);
  int tilesN = Co >> 4;
  int tilesM = (N + 31) >> 5;                 // 32 rows per wave
  if (wave >= tilesM * tilesN) return;        // uniform per wave
  int tm = (wave / tilesN) << 5;
  int tn = (wave % tilesN) << 4;
  int lane = threadIdx.x & 31;
  int grp  = lane >> 4;                        // 0 or 1
  int lan  = lane & 15;

  int r0 = tm + lan;      if (r0 >= N) r0 = N - 1;   // clamp (EXEC stays full)
  int r1 = tm + 16 + lan; if (r1 >= N) r1 = N - 1;
  long x0 = rowidx ? (long)rowidx[r0] : (long)r0;
  long x1 = rowidx ? (long)rowidx[r1] : (long)r1;
  const float* xp0 = X + x0 * Ci;
  const float* xp1 = X + x1 * Ci;
  const float* wp  = W + tn + lan;

  v8f c0 = {}, c1 = {};
  for (int kb = 0; kb < Ci; kb += 32) {
    int base = kb + 8 * grp;
    // A frag (16x32 f16 layout): lane needs K runs [base, base+8) and
    // [base+16, base+24) -> four float4 loads per fragment.
    v4f a0p0 = *(const v4f*)(xp0 + base);
    v4f a0p1 = *(const v4f*)(xp0 + base + 4);
    v4f a0p2 = *(const v4f*)(xp0 + base + 16);
    v4f a0p3 = *(const v4f*)(xp0 + base + 20);
    v4f a1p0 = *(const v4f*)(xp1 + base);
    v4f a1p1 = *(const v4f*)(xp1 + base + 4);
    v4f a1p2 = *(const v4f*)(xp1 + base + 16);
    v4f a1p3 = *(const v4f*)(xp1 + base + 20);
    v16h a0, a1, b;
#pragma unroll
    for (int h = 0; h < 4; ++h) {
      a0[h]      = (_Float16)a0p0[h];
      a0[h + 4]  = (_Float16)a0p1[h];
      a0[h + 8]  = (_Float16)a0p2[h];
      a0[h + 12] = (_Float16)a0p3[h];
      a1[h]      = (_Float16)a1p0[h];
      a1[h + 4]  = (_Float16)a1p1[h];
      a1[h + 8]  = (_Float16)a1p2[h];
      a1[h + 12] = (_Float16)a1p3[h];
    }
    // B frag (32x16 f16): K = kb + 8*grp + (h&7) + 16*(h>>3), matching A; N=lan
#pragma unroll
    for (int h = 0; h < 16; ++h)
      b[h] = (_Float16)wp[(long)(kb + 8 * grp + (h & 7) + ((h >> 3) << 4)) * Co];
    c0 = __builtin_amdgcn_wmma_f32_16x16x32_f16(false, a0, false, b,
                                                (short)0, c0, false, false);
    c1 = __builtin_amdgcn_wmma_f32_16x16x32_f16(false, a1, false, b,
                                                (short)0, c1, false, false);
  }

  int col = tn + lan;
  float bv = bias ? bias[col] : 0.0f;
#pragma unroll
  for (int r = 0; r < 8; ++r) {
    int row = tm + r + 8 * grp;                // C/D layout: M = r + 8*grp
    if (row < N) {
      float v = c0[r] + bv;
      if (act == 1)      v = (v > 0.0f) ? v : 0.01f * v;
      else if (act == 2) v = (v > 0.0f) ? v : 0.0f;
      long o = (long)row * Co + col;
      if (accumulate) Y[o] += v; else Y[o] = v;
    }
    int row1 = row + 16;
    if (row1 < N) {
      float v = c1[r] + bv;
      if (act == 1)      v = (v > 0.0f) ? v : 0.01f * v;
      else if (act == 2) v = (v > 0.0f) ? v : 0.0f;
      long o = (long)row1 * Co + col;
      if (accumulate) Y[o] += v; else Y[o] = v;
    }
  }
}

// ---------------- per-node attention logits: zl = z.al, zr = z.ar ----------------
__global__ void rowdot2(const float* __restrict__ z, const float* __restrict__ al,
                        const float* __restrict__ ar, float* __restrict__ zl,
                        float* __restrict__ zr, int N) {
  int i = blockIdx.x * blockDim.x + threadIdx.x;
  if (i >= N) return;
  const float* p = z + (long)i * HID;
  float s0 = 0.0f, s1 = 0.0f;
#pragma unroll 8
  for (int k = 0; k < HID; ++k) { float v = p[k]; s0 += v * al[k]; s1 += v * ar[k]; }
  zl[i] = s0; zr[i] = s1;
}

// order-preserving float <-> uint encode for atomic max
__device__ __forceinline__ unsigned fenc(float f) {
  unsigned u = __float_as_uint(f);
  return (u >> 31) ? ~u : (u | 0x80000000u);
}
__device__ __forceinline__ float fdec(unsigned u) {
  return __uint_as_float((u >> 31) ? (u & 0x7fffffffu) : ~u);
}
#define ENC_NEG_INF 0x007FFFFFu   // fenc(-inf)

// ---------------- edge softmax passes ----------------
__global__ void edge_score(const int* __restrict__ src, const int* __restrict__ dst,
                           const float* __restrict__ zl, const float* __restrict__ zr,
                           float* __restrict__ e, unsigned* __restrict__ menc, int E) {
  int i = blockIdx.x * blockDim.x + threadIdx.x;
  if (i >= E) return;
  float v = zl[src[i]] + zr[dst[i]];
  v = (v > 0.0f) ? v : 0.2f * v;               // leaky_relu(0.2)
  e[i] = v;
  atomicMax(menc + dst[i], fenc(v));
}

__global__ void edge_exp(const int* __restrict__ dst, float* __restrict__ e,
                         const unsigned* __restrict__ menc, float* __restrict__ denom, int E) {
  int i = blockIdx.x * blockDim.x + threadIdx.x;
  if (i >= E) return;
  int d = dst[i];
  float ex = __expf(e[i] - fdec(menc[d]));
  e[i] = ex;
  atomicAdd(denom + d, ex);
}

// each thread handles one (edge, dim-pair): 2 f32 atomic adds
__global__ void edge_agg(const int* __restrict__ src, const int* __restrict__ dst,
                         const float* __restrict__ ex, const float* __restrict__ denom,
                         const float* __restrict__ z, float* __restrict__ out, int E) {
  long t = (long)blockIdx.x * blockDim.x + threadIdx.x;
  if (t >= (long)E * 32) return;
  int ei = (int)(t >> 5);
  int d0 = (int)(t & 31) * 2;
  int s = src[ei], d = dst[ei];
  float a = ex[ei] / (denom[d] + 1e-9f);
  const float* zp = z + (long)s * HID + d0;
  float* op = out + (long)d * HID + d0;
  atomicAdd(op,     a * zp[0]);
  atomicAdd(op + 1, a * zp[1]);
}

// y = act(a) + x  (GAT residual / distGCN finish)
__global__ void ew_act_add(const float* __restrict__ a, const float* __restrict__ x,
                           float* __restrict__ y, long n, int act) {
  long i = (long)blockIdx.x * blockDim.x + threadIdx.x;
  if (i >= n) return;
  float v = a[i];
  if (act == 1)      v = (v > 0.0f) ? v : 0.01f * v;
  else if (act == 2) v = (v > 0.0f) ? v : 0.0f;
  y[i] = v + x[i];
}

// out[dst[e]*dim + d] += scale(e) * in[src_or_e * dim + d]
__global__ void scatter_rows(const float* __restrict__ in, const int* __restrict__ src,
                             const int* __restrict__ dst, const int* __restrict__ dist,
                             float alpha, float* __restrict__ out, int E, int dim) {
  long t = (long)blockIdx.x * blockDim.x + threadIdx.x;
  if (t >= (long)E * dim) return;
  int ei = (int)(t / dim);
  int d  = (int)(t % dim);
  float sc = dist ? __powf(alpha, (float)dist[ei]) : 1.0f;
  long s = src ? (long)src[ei] : (long)ei;
  atomicAdd(out + (long)dst[ei] * dim + d, sc * in[s * dim + d]);
}

// ---------------- batch-norm stats (C = 32): LDS partials + global atomics ----------------
__global__ void col_stats(const float* __restrict__ X, int N, float* __restrict__ sums) {
  __shared__ float ls[64];                     // [0:32) sum, [32:64) sumsq
  int tid = threadIdx.x;
  if (tid < 64) ls[tid] = 0.0f;
  __syncthreads();
  for (long r = (long)blockIdx.x * blockDim.x + tid; r < N;
       r += (long)gridDim.x * blockDim.x) {
    const float* p = X + r * 32;
#pragma unroll
    for (int c = 0; c < 32; ++c) {
      float v = p[c];
      atomicAdd(&ls[c], v);
      atomicAdd(&ls[32 + c], v * v);
    }
  }
  __syncthreads();
  if (tid < 64) atomicAdd(&sums[tid], ls[tid]);
}

// stats layout: [0:32) sum, [32:64) sumsq, [64:96) mean, [96:128) inv-std
__global__ void bn_finalize(float* __restrict__ stats, float invN) {
  int c = threadIdx.x;
  if (c >= 32) return;
  float mean = stats[c] * invN;
  float var  = stats[32 + c] * invN - mean * mean;
  stats[64 + c] = mean;
  stats[96 + c] = rsqrtf(var + 1e-5f);
}

// per row: relu(BN(t)) . W2 + b2 -> tanh (mode 1) or 2*sigmoid (mode 2)
__global__ void bn_rowdot(const float* __restrict__ T, const float* __restrict__ stats,
                          const float* __restrict__ g, const float* __restrict__ beta,
                          const float* __restrict__ W2, const float* __restrict__ b2,
                          float* __restrict__ out, int N, int mode) {
  int r = blockIdx.x * blockDim.x + threadIdx.x;
  if (r >= N) return;
  const float* p = T + (long)r * 32;
  float acc = 0.0f;
#pragma unroll
  for (int c = 0; c < 32; ++c) {
    float y = (p[c] - stats[64 + c]) * stats[96 + c] * g[c] + beta[c];
    y = (y > 0.0f) ? y : 0.0f;
    acc += y * W2[c];
  }
  acc += b2[0];
  out[r] = (mode == 1) ? tanhf(acc) : 2.0f / (1.0f + __expf(-acc));
}

// y[r] = X[r,:K] . w + b[0]
__global__ void rowdot_bias(const float* __restrict__ X, const float* __restrict__ w,
                            const float* __restrict__ b, float* __restrict__ y,
                            int N, int K) {
  int r = blockIdx.x * blockDim.x + threadIdx.x;
  if (r >= N) return;
  const float* p = X + (long)r * K;
  float acc = 0.0f;
  for (int k = 0; k < K; ++k) acc += p[k] * w[k];
  y[r] = acc + b[0];
}

__global__ void se_scatter(const float* __restrict__ s1, const float* __restrict__ s2,
                           const int* __restrict__ dseg, float* __restrict__ sumse, int N) {
  int r = blockIdx.x * blockDim.x + threadIdx.x;
  if (r >= N) return;
  atomicAdd(&sumse[dseg[r]], s1[r] * s2[r]);
}

__global__ void add2(const float* __restrict__ a, const float* __restrict__ b,
                     float* __restrict__ out, int n) {
  int i = blockIdx.x * blockDim.x + threadIdx.x;
  if (i >= n) return;
  out[i] = a[i] + b[i];
}

// ---------------------------------------------------------------------------
static inline unsigned gdiv(long n, int b) { return (unsigned)((n + b - 1) / b); }

static void run_gemm(const float* X, const int* rowidx, const float* W, const float* bias,
                     float* Y, int N, int Ci, int Co, int acc, int act, hipStream_t s) {
  long tiles = (long)((N + 31) / 32) * (Co / 16);
  gemm_wmma<<<gdiv(tiles, 8), 256, 0, s>>>(X, rowidx, W, bias, Y, N, Ci, Co, acc, act);
}

static void run_gat(float* x, float* z, float* agg, float* zl, float* zr, float* e,
                    unsigned* menc, float* denom, const float* W, const float* al,
                    const float* ar, const int* src, const int* dst, int N, int E,
                    int act, hipStream_t s) {
  run_gemm(x, nullptr, W, nullptr, z, N, HID, HID, 0, 0, s);
  rowdot2<<<gdiv(N, 256), 256, 0, s>>>(z, al, ar, zl, zr, N);
  fill_u32<<<gdiv(N, 256), 256, 0, s>>>(menc, ENC_NEG_INF, N);
  fill_u32<<<gdiv(N, 256), 256, 0, s>>>((unsigned*)denom, 0u, N);
  fill_u32<<<gdiv((long)N * HID, 256), 256, 0, s>>>((unsigned*)agg, 0u, (long)N * HID);
  edge_score<<<gdiv(E, 256), 256, 0, s>>>(src, dst, zl, zr, e, menc, E);
  edge_exp<<<gdiv(E, 256), 256, 0, s>>>(dst, e, menc, denom, E);
  edge_agg<<<gdiv((long)E * 32, 256), 256, 0, s>>>(src, dst, e, denom, z, agg, E);
  ew_act_add<<<gdiv((long)N * HID, 256), 256, 0, s>>>(agg, x, x, (long)N * HID, act);
}

extern "C" void kernel_launch(void* const* d_in, const int* in_sizes, int n_in,
                              void* d_out, int out_size, void* d_ws, size_t ws_size,
                              hipStream_t stream) {
  // ---- inputs (setup_inputs dict order, params flattened in literal order) ----
  const float* r_node      = (const float*)d_in[0];
  const float* r_node_elu1 = (const float*)d_in[2];
  const float* W_emb_r    = (const float*)d_in[4];
  const float* W_emb_i    = (const float*)d_in[5];
  // d_in[6] = W_emb_e : edge embedding is computed-but-unused in reference -> skipped
  const float* W_emb_solv = (const float*)d_in[7];
  const float* r2r_W  = (const float*)d_in[8];
  const float* r2r_al = (const float*)d_in[9];
  const float* r2r_ar = (const float*)d_in[10];
  const float* i2i_W  = (const float*)d_in[11];
  const float* i2i_al = (const float*)d_in[12];
  const float* i2i_ar = (const float*)d_in[13];
  const float* solv_W  = (const float*)d_in[14];
  const float* solv_al = (const float*)d_in[15];
  const float* solv_ar = (const float*)d_in[16];
  const float* d2d_W   = (const float*)d_in[17];
  const float* se1_W1 = (const float*)d_in[18];
  const float* se1_b1 = (const float*)d_in[19];
  const float* se1_g  = (const float*)d_in[20];
  const float* se1_be = (const float*)d_in[21];
  const float* se1_W2 = (const float*)d_in[22];
  const float* se1_b2 = (const float*)d_in[23];
  const float* se2_W1 = (const float*)d_in[24];
  const float* se2_b1 = (const float*)d_in[25];
  const float* se2_g  = (const float*)d_in[26];
  const float* se2_be = (const float*)d_in[27];
  const float* se2_W2 = (const float*)d_in[28];
  const float* se2_b2 = (const float*)d_in[29];
  const float* ref_W1 = (const float*)d_in[30];
  const float* ref_b1 = (const float*)d_in[31];
  const float* ref_W2 = (const float*)d_in[32];
  const float* ref_b2 = (const float*)d_in[33];
  const float* ref_W3 = (const float*)d_in[34];
  const float* ref_b3 = (const float*)d_in[35];
  const int* r2r_src = (const int*)d_in[36];
  const int* r2r_dst = (const int*)d_in[37];
  const int* i2i_src = (const int*)d_in[38];
  const int* i2i_dst = (const int*)d_in[39];
  const int* i2d_src = (const int*)d_in[40];
  const int* i2d_dst = (const int*)d_in[41];
  const int* d2d_src = (const int*)d_in[42];
  const int* d2d_dst = (const int*)d_in[43];
  const int* d2d_dist = (const int*)d_in[44];
  const int* r_seg = (const int*)d_in[45];
  const int* d_seg = (const int*)d_in[46];
  const int* s1_src = (const int*)d_in[47];
  const int* s1_dst = (const int*)d_in[48];
  const int* s1_seg = (const int*)d_in[49];
  float* out = (float*)d_out;

  // ---- workspace layout ----
  char* wp = (char*)d_ws;
  auto alloc = [&](size_t bytes) -> void* {
    void* p = (void*)wp;
    wp += (bytes + 255) & ~(size_t)255;
    return p;
  };
  float* r2    = (float*)alloc((size_t)NRn * HID * 4);
  float* iF    = (float*)alloc((size_t)NRn * HID * 4);
  float* sF    = (float*)alloc((size_t)NSn * HID * 4);
  float* z     = (float*)alloc((size_t)NRn * HID * 4);  // reused as t1/t2 later
  float* agg   = (float*)alloc((size_t)NRn * HID * 4);
  float* d1    = (float*)alloc((size_t)NDn * HID * 4);
  float* d2    = (float*)alloc((size_t)NDn * HID * 4);
  float* zl    = (float*)alloc((size_t)NRn * 4);
  float* zr    = (float*)alloc((size_t)NRn * 4);
  float* ebuf  = (float*)alloc((size_t)ERn * 4);
  unsigned* menc = (unsigned*)alloc((size_t)NRn * 4);
  float* denom = (float*)alloc((size_t)NRn * 4);
  float* h     = (float*)alloc((size_t)Bn * HID * 4);
  float* solv  = (float*)alloc((size_t)Bn * HID * 4);
  float* rp1   = (float*)alloc((size_t)Bn * 32 * 4);
  float* rp2   = (float*)alloc((size_t)Bn * 16 * 4);
  float* referp = (float*)alloc((size_t)Bn * 4);
  float* se1v  = (float*)alloc((size_t)NDn * 4);
  float* se2v  = (float*)alloc((size_t)NDn * 4);
  float* sumse = (float*)alloc((size_t)Bn * 4);
  float* stats = (float*)alloc(128 * 4);
  (void)ws_size; (void)n_in; (void)in_sizes; (void)out_size;

  // ---- embeddings (WMMA): NODE_DIM=128 -> HID=64 ----
  run_gemm(r_node, nullptr, W_emb_r, nullptr, r2, NRn, 128, HID, 0, 0, stream);
  run_gemm(r_node, nullptr, W_emb_i, nullptr, iF, NRn, 128, HID, 0, 0, stream);
  run_gemm(r_node_elu1, nullptr, W_emb_solv, nullptr, sF, NSn, 128, HID, 0, 0, stream);

  // ---- GAT stacks ----
  for (int l = 0; l < NLn; ++l) {
    run_gat(r2, z, agg, zl, zr, ebuf, menc, denom,
            r2r_W + (size_t)l * HID * HID, r2r_al + (size_t)l * HID,
            r2r_ar + (size_t)l * HID, r2r_src, r2r_dst, NRn, ERn, /*lrelu*/1, stream);
    run_gat(iF, z, agg, zl, zr, ebuf, menc, denom,
            i2i_W + (size_t)l * HID * HID, i2i_al + (size_t)l * HID,
            i2i_ar + (size_t)l * HID, i2i_src, i2i_dst, NRn, EIn, 1, stream);
    run_gat(sF, z, agg, zl, zr, ebuf, menc, denom,
            solv_W + (size_t)l * HID * HID, solv_al + (size_t)l * HID,
            solv_ar + (size_t)l * HID, s1_src, s1_dst, NSn, ESn, /*relu*/2, stream);
  }

  // ---- i2d: d1 = segment_sum(i[i2d_src]) ----
  fill_u32<<<gdiv((long)NDn * HID, 256), 256, 0, stream>>>((unsigned*)d1, 0u, (long)NDn * HID);
  scatter_rows<<<gdiv((long)EIDn * HID, 256), 256, 0, stream>>>(
      iF, i2d_src, i2d_dst, nullptr, 1.0f, d1, EIDn, HID);

  // ---- distGCN: agg = segsum(alpha^dist * d1[src]); d2 = lrelu(agg@W)+d1 ----
  fill_u32<<<gdiv((long)NDn * HID, 256), 256, 0, stream>>>((unsigned*)agg, 0u, (long)NDn * HID);
  scatter_rows<<<gdiv((long)EDn * HID, 256), 256, 0, stream>>>(
      d1, d2d_src, d2d_dst, d2d_dist, 0.1f, agg, EDn, HID);
  run_gemm(agg, nullptr, d2d_W, nullptr, z, NDn, HID, HID, 0, 0, stream);
  ew_act_add<<<gdiv((long)NDn * HID, 256), 256, 0, stream>>>(z, d1, d2, (long)NDn * HID, 1);

  // ---- graph pooling h = segsum(r2, r_seg); solv = segsum(s, s1_seg) ----
  fill_u32<<<gdiv((long)Bn * HID, 256), 256, 0, stream>>>((unsigned*)h, 0u, (long)Bn * HID);
  scatter_rows<<<gdiv((long)NRn * HID, 256), 256, 0, stream>>>(
      r2, nullptr, r_seg, nullptr, 1.0f, h, NRn, HID);
  fill_u32<<<gdiv((long)Bn * HID, 256), 256, 0, stream>>>((unsigned*)solv, 0u, (long)Bn * HID);
  scatter_rows<<<gdiv((long)NSn * HID, 256), 256, 0, stream>>>(
      sF, nullptr, s1_seg, nullptr, 1.0f, solv, NSn, HID);
  // solv = 0.5*h1 + 0.5*h2 with h2 == h1 -> solv unchanged

  // ---- refer_p MLP: 64 -> 32 -> 16 -> 1 ----
  run_gemm(h,   nullptr, ref_W1, ref_b1, rp1, Bn, HID, 32, 0, 1, stream);
  run_gemm(rp1, nullptr, ref_W2, ref_b2, rp2, Bn, 32, 16, 0, 1, stream);
  rowdot_bias<<<gdiv(Bn, 256), 256, 0, stream>>>(rp2, ref_W3, ref_b3, referp, Bn, 16);

  // ---- se1: t1 = concat(d1, solv[d_seg]) @ W1 + b1 (split-K, no materialized concat) ----
  float* t = z;  // z region free now: reuse for ND x 32 activations
  run_gemm(d1,   nullptr, se1_W1,            nullptr, t, NDn, HID, 32, 0, 0, stream);
  run_gemm(solv, d_seg,   se1_W1 + 64 * 32,  se1_b1,  t, NDn, HID, 32, 1, 0, stream);
  fill_u32<<<1, 128, 0, stream>>>((unsigned*)stats, 0u, 128);
  col_stats<<<512, 256, 0, stream>>>(t, NDn, stats);
  bn_finalize<<<1, 32, 0, stream>>>(stats, 1.0f / (float)NDn);
  bn_rowdot<<<gdiv(NDn, 256), 256, 0, stream>>>(t, stats, se1_g, se1_be, se1_W2, se1_b2,
                                                se1v, NDn, /*tanh*/1);

  // ---- se2: t2 = concat(d2, h[d_seg], solv[d_seg]) @ W1 + b1 ----
  run_gemm(d2,   nullptr, se2_W1,             nullptr, t, NDn, HID, 32, 0, 0, stream);
  run_gemm(h,    d_seg,   se2_W1 + 64 * 32,   nullptr, t, NDn, HID, 32, 1, 0, stream);
  run_gemm(solv, d_seg,   se2_W1 + 128 * 32,  se2_b1,  t, NDn, HID, 32, 1, 0, stream);
  fill_u32<<<1, 128, 0, stream>>>((unsigned*)stats, 0u, 128);
  col_stats<<<512, 256, 0, stream>>>(t, NDn, stats);
  bn_finalize<<<1, 32, 0, stream>>>(stats, 1.0f / (float)NDn);
  bn_rowdot<<<gdiv(NDn, 256), 256, 0, stream>>>(t, stats, se2_g, se2_be, se2_W2, se2_b2,
                                                se2v, NDn, /*2*sigmoid*/2);

  // ---- se = se1*se2; sum over d_seg; out = refer_p + sum_se ----
  fill_u32<<<gdiv(Bn, 256), 256, 0, stream>>>((unsigned*)sumse, 0u, Bn);
  se_scatter<<<gdiv(NDn, 256), 256, 0, stream>>>(se1v, se2v, d_seg, sumse, NDn);
  add2<<<gdiv(Bn, 256), 256, 0, stream>>>(referp, sumse, out, Bn);
}
